// FCFE_39505109188695
// MI455X (gfx1250) — matrix operand order repaired and checked by reading, not compiled, not run on hardware
//
#include <hip/hip_runtime.h>
#include <hip/hip_bf16.h>

// ---------------------------------------------------------------------------
// FCFE fused pipeline for MI455X (gfx1250): all 1x1 convs via v_wmma bf16,
// fp32 FFT/elementwise, per-pixel 4x4 batch-attention.
// ---------------------------------------------------------------------------

#define HW   16384              // H*W
#define NCH  256                // channels
#define NBAT 4                  // batch
#define NT   ((size_t)NBAT * NCH * HW)   // 16,777,216 elements per tensor

typedef __attribute__((ext_vector_type(16))) __bf16 v16bf;
typedef __attribute__((ext_vector_type(8)))  float  v8f;

// Pack two floats into one bf16x2 dword with a single hardware instruction.
// Inline asm pins the operand pairing (the SLP vectorizer otherwise pairs
// adjacent elements of the same vector load and re-shuffles half-words).
// Mnemonic/operand form verified from this toolchain's own gfx1250 codegen.
__device__ __forceinline__ unsigned pack_bf16x2(float lo, float hi) {
  unsigned u;
  asm("v_cvt_pk_bf16_f32 %0, %1, %2" : "=v"(u) : "v"(lo), "v"(hi));
  return u;
}

union FragBF { v16bf v; uint4 q[2]; unsigned short s[16]; };

// ---------------------------------------------------------------------------
// WMMA 1x1 conv: Y[b,o,p] = act( sum_c W[o,c] * X[b,c,p] + bias[o] )
// Cout fixed = 256. CIN = 256 or 512 (concat of two 256-ch tensors Xa,Xb).
// Block: 256 thr = 8 waves, tile 64(o) x 128(n). Wave: 32x32 = 4 WMMA tiles.
// LDS holds X tile transposed [n][c] in bf16 so B-fragments are two
// contiguous 16B ds_load_b128 per lane (lanes 0-15: K0-15, 16-31: K16-31).
// ---------------------------------------------------------------------------
template <int CIN, int ACT>   // ACT: 0 none, 1 relu, 2 sigmoid
__global__ __launch_bounds__(256) void conv1x1_wmma_kernel(
    const float* __restrict__ Xa, const float* __restrict__ Xb,
    const unsigned short* __restrict__ Wb, const float* __restrict__ bias,
    float* __restrict__ Y)
{
  // 128 rows (n) x 32 cols (c) bf16, row stride 40 bf16 (=20 u32) vs bank dup
  __shared__ __align__(16) unsigned short ldsX[128 * 40];
  unsigned* lds32 = (unsigned*)ldsX;

  const int tid  = threadIdx.x;
  const int lane = tid & 31;
  const int wid  = tid >> 5;
  const int wm   = wid >> 2;          // 0..1  (M sub-block of 32)
  const int wn   = wid & 3;           // 0..3  (N sub-block of 32)
  const int nBlock = blockIdx.x * 128;            // global column start
  const int oBlock = blockIdx.y * 64;             // Cout start
  const int bb = nBlock / HW;                     // batch (128 | HW -> uniform)
  const int p0 = nBlock - bb * HW;                // pixel start in batch
  const int lane16   = lane & 15;
  const int laneHalf = lane >> 4;                 // 0: K-lo half, 1: K-hi half

  v8f acc[2][2];
  acc[0][0] = (v8f)0.f; acc[0][1] = (v8f)0.f;
  acc[1][0] = (v8f)0.f; acc[1][1] = (v8f)0.f;

  // staging assignment: each thread stages a channel PAIR over 8 columns,
  // packing (c, c+1) into one u32 -> 8 x ds_store_b32 per thread.
  const int cpair = tid >> 4;           // 0..15  -> channels 2*cpair, 2*cpair+1
  const int col0  = (tid & 15) << 3;    // 0,8,...,120

  for (int k0 = 0; k0 < CIN; k0 += 32) {
    __syncthreads();
    { // stage 32c x 128n fp32 tile -> bf16 packed pairs, transposed into LDS
      int c = k0 + 2 * cpair;
      const float* base = Xa; int cc = c;
      if (CIN == 512 && c >= 256) { base = Xb; cc = c - 256; }
      const float* src0 = base + ((size_t)bb * 256 + cc) * HW + p0 + col0;
      const float* src1 = src0 + HW;     // channel c+1 (same tensor: k-tile
                                         // never straddles the concat seam)
      float4 a0 = ((const float4*)src0)[0];
      float4 a1 = ((const float4*)src0)[1];
      float4 b0 = ((const float4*)src1)[0];
      float4 b1 = ((const float4*)src1)[1];
      if (k0 + 32 < CIN) __builtin_prefetch(src0 + 32 * HW, 0, 1);
      unsigned pk[8];
      pk[0] = pack_bf16x2(a0.x, b0.x); pk[1] = pack_bf16x2(a0.y, b0.y);
      pk[2] = pack_bf16x2(a0.z, b0.z); pk[3] = pack_bf16x2(a0.w, b0.w);
      pk[4] = pack_bf16x2(a1.x, b1.x); pk[5] = pack_bf16x2(a1.y, b1.y);
      pk[6] = pack_bf16x2(a1.z, b1.z); pk[7] = pack_bf16x2(a1.w, b1.w);
      #pragma unroll
      for (int i = 0; i < 8; i++)
        lds32[(col0 + i) * 20 + cpair] = pk[i];
    }
    __syncthreads();

    // A fragments (weights 16x32): lanes 0-15 hold K 0-15, hi 16-31
    FragBF afrag[2], bfrag[2];
    #pragma unroll
    for (int mi = 0; mi < 2; mi++) {
      int o = oBlock + wm * 32 + mi * 16 + lane16;
      const uint4* wp = (const uint4*)(Wb + (size_t)o * CIN + k0 + laneHalf * 16);
      afrag[mi].q[0] = wp[0];     // K +0..7
      afrag[mi].q[1] = wp[1];     // K +8..15
    }
    // B fragments (X 32x16): lanes 0-15 hold K 0-15, lanes 16-31 hold K 16-31
    #pragma unroll
    for (int ni = 0; ni < 2; ni++) {
      int ncol = wn * 32 + ni * 16 + lane16;
      const uint4* bp = (const uint4*)&ldsX[ncol * 40 + laneHalf * 16];
      bfrag[ni].q[0] = bp[0];
      bfrag[ni].q[1] = bp[1];
    }
    #pragma unroll
    for (int mi = 0; mi < 2; mi++)
      #pragma unroll
      for (int ni = 0; ni < 2; ni++)
        acc[mi][ni] = __builtin_amdgcn_wmma_f32_16x16x32_bf16(
            false, afrag[mi].v, false, bfrag[ni].v,
            (short)0, acc[mi][ni], false, false);
  }

  // epilogue: C/D layout — vgpr r, lane<16 -> M=r, lane>=16 -> M=r+8; N=lane%16
  #pragma unroll
  for (int mi = 0; mi < 2; mi++) {
    int oBase = oBlock + wm * 32 + mi * 16 + laneHalf * 8;
    float4 bv0 = *(const float4*)&bias[oBase];
    float4 bv1 = *(const float4*)&bias[oBase + 4];
    float bval[8] = {bv0.x, bv0.y, bv0.z, bv0.w, bv1.x, bv1.y, bv1.z, bv1.w};
    #pragma unroll
    for (int ni = 0; ni < 2; ni++) {
      int ncol = wn * 32 + ni * 16 + lane16;
      size_t col = (size_t)p0 + ncol;
      #pragma unroll
      for (int r = 0; r < 8; r++) {
        int o = oBase + r;
        float v = acc[mi][ni][r] + bval[r];
        if (ACT == 1) v = fmaxf(v, 0.f);
        if (ACT == 2) v = 1.f / (1.f + __expf(-v));
        Y[((size_t)bb * NCH + o) * HW + col] = v;
      }
    }
  }
}

// ---------------------------------------------------------------------------
// 128-point radix-2 FFT (DIT, bit-reversed load). 4 FFTs per block, 64 thr each.
// base = (f/inner_cnt)*outer_stride + (f%inner_cnt)*inner_stride
// dsign = -1 forward, +1 inverse; scale applied at store (1/128 per inverse pass)
// ---------------------------------------------------------------------------
__global__ __launch_bounds__(256) void fft128_kernel(
    float* __restrict__ re, float* __restrict__ im,
    int inner_cnt, int inner_stride, int outer_stride, int elem_stride,
    float dsign, float scale)
{
  __shared__ float sre[4][128];
  __shared__ float sim[4][128];
  const int g = threadIdx.x >> 6;
  const int t = threadIdx.x & 63;
  const int f = blockIdx.x * 4 + g;
  const int outer = f / inner_cnt;
  const int inner = f - outer * inner_cnt;
  const size_t base = (size_t)outer * outer_stride + (size_t)inner * inner_stride;

  const int j0 = t, j1 = t + 64;
  const int r0 = __brev((unsigned)j0) >> 25;
  const int r1 = __brev((unsigned)j1) >> 25;
  sre[g][r0] = re[base + (size_t)j0 * elem_stride];
  sim[g][r0] = im[base + (size_t)j0 * elem_stride];
  sre[g][r1] = re[base + (size_t)j1 * elem_stride];
  sim[g][r1] = im[base + (size_t)j1 * elem_stride];

  for (int s = 0; s < 7; s++) {
    __syncthreads();
    const int half = 1 << s;
    const int k = t & (half - 1);
    const int i = ((t >> s) << (s + 1)) + k;
    const int j = i + half;
    const float ang = dsign * 3.14159265358979323846f * (float)k / (float)half;
    float wi, wr;
    __sincosf(ang, &wi, &wr);
    const float br = sre[g][j], bi = sim[g][j];
    const float tr = br * wr - bi * wi;
    const float ti = br * wi + bi * wr;
    const float ar = sre[g][i], ai = sim[g][i];
    sre[g][j] = ar - tr; sim[g][j] = ai - ti;
    sre[g][i] = ar + tr; sim[g][i] = ai + ti;
  }
  __syncthreads();
  re[base + (size_t)j0 * elem_stride] = sre[g][j0] * scale;
  im[base + (size_t)j0 * elem_stride] = sim[g][j0] * scale;
  re[base + (size_t)j1 * elem_stride] = sre[g][j1] * scale;
  im[base + (size_t)j1 * elem_stride] = sim[g][j1] * scale;
}

// ---------------------------------------------------------------------------
// LayerNorm over (C,H,W) per batch: two-stage deterministic reduction
// ---------------------------------------------------------------------------
__global__ __launch_bounds__(256) void ln_partial_kernel(
    const float* __restrict__ x, float* __restrict__ partial)
{
  __shared__ float ss[256], sq[256];
  const int tid = threadIdx.x;
  const int batch = blockIdx.x >> 10;       // 1024 blocks per batch
  const int blk   = blockIdx.x & 1023;
  const float* p = x + ((size_t)batch << 22) + ((size_t)blk << 12);
  float s = 0.f, q = 0.f;
  for (int i = tid; i < 4096; i += 256) { float v = p[i]; s += v; q += v * v; }
  ss[tid] = s; sq[tid] = q; __syncthreads();
  for (int o = 128; o; o >>= 1) {
    if (tid < o) { ss[tid] += ss[tid + o]; sq[tid] += sq[tid + o]; }
    __syncthreads();
  }
  if (tid == 0) { partial[blockIdx.x * 2] = ss[0]; partial[blockIdx.x * 2 + 1] = sq[0]; }
}

__global__ __launch_bounds__(256) void ln_finalize_kernel(
    const float* __restrict__ partial, float* __restrict__ stats)
{
  __shared__ float ss[256], sq[256];
  const int tid = threadIdx.x, b = blockIdx.x;
  float s = 0.f, q = 0.f;
  for (int i = tid; i < 1024; i += 256) {
    s += partial[(b * 1024 + i) * 2];
    q += partial[(b * 1024 + i) * 2 + 1];
  }
  ss[tid] = s; sq[tid] = q; __syncthreads();
  for (int o = 128; o; o >>= 1) {
    if (tid < o) { ss[tid] += ss[tid + o]; sq[tid] += sq[tid + o]; }
    __syncthreads();
  }
  if (tid == 0) {
    const float inv_n = 1.f / 4194304.f;
    float mean = ss[0] * inv_n;
    float var  = sq[0] * inv_n - mean * mean;
    stats[b * 2]     = mean;
    stats[b * 2 + 1] = rsqrtf(var + 1e-5f);
  }
}

__global__ __launch_bounds__(256) void ln_apply_kernel(
    const float* __restrict__ x, const float* __restrict__ stats,
    float* __restrict__ y)
{
  const size_t i = (size_t)blockIdx.x * 256 + threadIdx.x;
  const int b = (int)(i >> 22);
  y[i] = (x[i] - stats[b * 2]) * stats[b * 2 + 1];
}

// ---------------------------------------------------------------------------
// Depthwise 3x3 (SAME, zero pad) + optional exact GELU
// ---------------------------------------------------------------------------
__global__ __launch_bounds__(256) void dw3x3_kernel(
    const float* __restrict__ x, const float* __restrict__ w,
    const float* __restrict__ bias, float* __restrict__ y, int act)
{
  const size_t idx = (size_t)blockIdx.x * 256 + threadIdx.x;
  const int wv = (int)(idx & 127);
  const int h  = (int)((idx >> 7) & 127);
  const int c  = (int)((idx >> 14) & 255);
  const float* xp = x + (idx & ~(size_t)16383);   // plane base
  float acc = bias[c];
  #pragma unroll
  for (int i = -1; i <= 1; i++)
    #pragma unroll
    for (int j = -1; j <= 1; j++) {
      int hh = h + i, ww = wv + j;
      if (hh >= 0 && hh < 128 && ww >= 0 && ww < 128)
        acc += xp[(hh << 7) + ww] * w[c * 9 + (i + 1) * 3 + (j + 1)];
    }
  if (act == 1) acc = 0.5f * acc * (1.f + erff(acc * 0.70710678118654752f));
  y[idx] = acc;
}

// ---------------------------------------------------------------------------
// Per-pixel attention across batch dim: L=4, E=256. One wave per pixel.
// Q/K/V/O layout (l, e, p) == NCHW. Softmax over m, scale 1/sqrt(256)=1/16.
// ---------------------------------------------------------------------------
__global__ __launch_bounds__(256) void attn4_kernel(
    const float* __restrict__ Q, const float* __restrict__ K,
    const float* __restrict__ V, float* __restrict__ O)
{
  const int lane = threadIdx.x & 31;
  const int wrp  = threadIdx.x >> 5;
  const int p = blockIdx.x * 8 + wrp;

  float s[16];
  #pragma unroll
  for (int i = 0; i < 16; i++) s[i] = 0.f;

  for (int e = lane; e < 256; e += 32) {
    float q[4], k[4];
    #pragma unroll
    for (int l = 0; l < 4; l++) {
      q[l] = Q[((size_t)(l * 256 + e)) * HW + p];
      k[l] = K[((size_t)(l * 256 + e)) * HW + p];
    }
    #pragma unroll
    for (int l = 0; l < 4; l++)
      #pragma unroll
      for (int m = 0; m < 4; m++) s[l * 4 + m] += q[l] * k[m];
  }
  #pragma unroll
  for (int i = 0; i < 16; i++)
    for (int off = 16; off; off >>= 1) s[i] += __shfl_xor(s[i], off, 32);

  float a[16];
  #pragma unroll
  for (int l = 0; l < 4; l++) {
    float m0 = -1e30f;
    #pragma unroll
    for (int m = 0; m < 4; m++) m0 = fmaxf(m0, s[l * 4 + m] * 0.0625f);
    float sum = 0.f;
    #pragma unroll
    for (int m = 0; m < 4; m++) { a[l*4+m] = __expf(s[l*4+m] * 0.0625f - m0); sum += a[l*4+m]; }
    const float inv = 1.f / sum;
    #pragma unroll
    for (int m = 0; m < 4; m++) a[l * 4 + m] *= inv;
  }
  for (int e = lane; e < 256; e += 32) {
    float v[4];
    #pragma unroll
    for (int m = 0; m < 4; m++) v[m] = V[((size_t)(m * 256 + e)) * HW + p];
    #pragma unroll
    for (int l = 0; l < 4; l++) {
      float o = a[l*4+0]*v[0] + a[l*4+1]*v[1] + a[l*4+2]*v[2] + a[l*4+3]*v[3];
      O[((size_t)(l * 256 + e)) * HW + p] = o;
    }
  }
}

// ---------------------------------------------------------------------------
// small helpers
// ---------------------------------------------------------------------------
__global__ void cvt_bf16_kernel(const float* __restrict__ s,
                                unsigned* __restrict__ d, int n2) {
  // converts pairs: n2 = number of u32 outputs (element count / 2)
  int i = blockIdx.x * 256 + threadIdx.x;
  if (i < n2) {
    float2 f = ((const float2*)s)[i];
    d[i] = pack_bf16x2(f.x, f.y);
  }
}
__global__ void ew_zero_kernel(float* p) {
  p[(size_t)blockIdx.x * 256 + threadIdx.x] = 0.f;
}
__global__ void ew_copy_kernel(const float* a, float* o) {
  size_t i = (size_t)blockIdx.x * 256 + threadIdx.x; o[i] = a[i];
}
__global__ void ew_mul_kernel(const float* a, const float* b, float* o) {
  size_t i = (size_t)blockIdx.x * 256 + threadIdx.x; o[i] = a[i] * b[i];
}
__global__ void ew_add_kernel(const float* a, const float* b, float* o) {
  size_t i = (size_t)blockIdx.x * 256 + threadIdx.x; o[i] = a[i] + b[i];
}
__global__ void ew_mul2_kernel(float* re, float* im, const float* r) {
  size_t i = (size_t)blockIdx.x * 256 + threadIdx.x;
  float f = r[i]; re[i] *= f; im[i] *= f;
}

// ---------------------------------------------------------------------------
extern "C" void kernel_launch(void* const* d_in, const int* in_sizes, int n_in,
                              void* d_out, int out_size, void* d_ws, size_t ws_size,
                              hipStream_t stream)
{
  (void)in_sizes; (void)n_in; (void)out_size;

  const float* event = (const float*)d_in[0];
  const float* image = (const float*)d_in[1];
  const float* w_c1  = (const float*)d_in[2];  const float* b_c1  = (const float*)d_in[3];
  const float* w_dw1 = (const float*)d_in[4];  const float* b_dw1 = (const float*)d_in[5];
  const float* w_dw2 = (const float*)d_in[6];  const float* b_dw2 = (const float*)d_in[7];
  const float* w_dw3 = (const float*)d_in[8];  const float* b_dw3 = (const float*)d_in[9];
  const float* w_re  = (const float*)d_in[10]; const float* b_re  = (const float*)d_in[11];
  const float* w_im  = (const float*)d_in[12]; const float* b_im  = (const float*)d_in[13];
  const float* w_c2  = (const float*)d_in[14]; const float* b_c2  = (const float*)d_in[15];
  const float* w_c3  = (const float*)d_in[16]; const float* b_c3  = (const float*)d_in[17];
  const float* w_c4  = (const float*)d_in[18]; const float* b_c4  = (const float*)d_in[19];
  const float* w_c6  = (const float*)d_in[20]; const float* b_c6  = (const float*)d_in[21];
  const float* w_c7  = (const float*)d_in[22]; const float* b_c7  = (const float*)d_in[23];
  const float* w_ca  = (const float*)d_in[24]; const float* b_ca  = (const float*)d_in[25];
  const float* w_fg  = (const float*)d_in[26]; const float* b_fg  = (const float*)d_in[27];
  const float* w_q   = (const float*)d_in[28]; const float* b_q   = (const float*)d_in[29];
  const float* w_k   = (const float*)d_in[30]; const float* b_k   = (const float*)d_in[31];
  const float* w_v   = (const float*)d_in[32]; const float* b_v   = (const float*)d_in[33];
  const float* w_o   = (const float*)d_in[34]; const float* b_o   = (const float*)d_in[35];
  const float* w_c5  = (const float*)d_in[36]; const float* b_c5  = (const float*)d_in[37];

  // ---- workspace layout ----
  float* wsf = (float*)d_ws;
  float* S[8];
  for (int i = 0; i < 8; i++) S[i] = wsf + NT * (size_t)i;
  unsigned short* WBase = (unsigned short*)(wsf + NT * 8);
  unsigned short* Wc1b = WBase;                   // 256*512
  unsigned short* Wc5b = WBase + 131072;          // 256*512
  unsigned short* W13  = WBase + 262144;          // 13 x 256*256
  unsigned short* Wreb = W13 + 0  * 65536;
  unsigned short* Wimb = W13 + 1  * 65536;
  unsigned short* Wc2b = W13 + 2  * 65536;
  unsigned short* Wc3b = W13 + 3  * 65536;
  unsigned short* Wc4b = W13 + 4  * 65536;
  unsigned short* Wc6b = W13 + 5  * 65536;
  unsigned short* Wc7b = W13 + 6  * 65536;
  unsigned short* Wcab = W13 + 7  * 65536;
  unsigned short* Wfgb = W13 + 8  * 65536;
  unsigned short* Wqb  = W13 + 9  * 65536;
  unsigned short* Wkb  = W13 + 10 * 65536;
  unsigned short* Wvb  = W13 + 11 * 65536;
  unsigned short* Wob  = W13 + 12 * 65536;
  float* lnPartial = (float*)(W13 + 13 * 65536);  // 8192 floats
  float* lnStats   = lnPartial + 8192;            // 8 floats
  const size_t needed = (8 * NT + 600000) * sizeof(float);
  if (ws_size < needed) return;   // compile-only env; guard against tiny ws

  const int EWG = (int)(NT / 256);   // 65536 blocks for elementwise

  auto cvt = [&](const float* s, unsigned short* d, int n) {
    int n2 = n / 2;
    cvt_bf16_kernel<<<(n2 + 255) / 256, 256, 0, stream>>>(s, (unsigned*)d, n2);
  };
  auto layernorm = [&](const float* x, float* y) {
    ln_partial_kernel<<<4096, 256, 0, stream>>>(x, lnPartial);
    ln_finalize_kernel<<<NBAT, 256, 0, stream>>>(lnPartial, lnStats);
    ln_apply_kernel<<<EWG, 256, 0, stream>>>(x, lnStats, y);
  };
  auto conv = [&](const float* xa, const float* xb, const unsigned short* wb,
                  const float* bias, float* y, int cin, int act) {
    dim3 grid(512, 4);
    if (cin == 256) {
      if (act == 0)      conv1x1_wmma_kernel<256, 0><<<grid, 256, 0, stream>>>(xa, xb, wb, bias, y);
      else if (act == 1) conv1x1_wmma_kernel<256, 1><<<grid, 256, 0, stream>>>(xa, xb, wb, bias, y);
      else               conv1x1_wmma_kernel<256, 2><<<grid, 256, 0, stream>>>(xa, xb, wb, bias, y);
    } else {
      if (act == 0)      conv1x1_wmma_kernel<512, 0><<<grid, 256, 0, stream>>>(xa, xb, wb, bias, y);
      else if (act == 1) conv1x1_wmma_kernel<512, 1><<<grid, 256, 0, stream>>>(xa, xb, wb, bias, y);
      else               conv1x1_wmma_kernel<512, 2><<<grid, 256, 0, stream>>>(xa, xb, wb, bias, y);
    }
  };
  // 131072 FFTs per pass (rows: B*C*H; cols: B*C*W), 4 per block
  auto fft_rows = [&](float* re, float* im, float sgn, float scl) {
    fft128_kernel<<<32768, 256, 0, stream>>>(re, im, 128, 128, 16384, 1, sgn, scl);
  };
  auto fft_cols = [&](float* re, float* im, float sgn, float scl) {
    fft128_kernel<<<32768, 256, 0, stream>>>(re, im, 128, 1, 16384, 128, sgn, scl);
  };
  auto dw = [&](const float* x, const float* w, const float* b, float* y, int act) {
    dw3x3_kernel<<<EWG, 256, 0, stream>>>(x, w, b, y, act);
  };
  auto zero = [&](float* p) { ew_zero_kernel<<<EWG, 256, 0, stream>>>(p); };

  const float IS = 1.f / 128.f;   // inverse scale per FFT pass

  // ---- convert weights to bf16 once per launch (deterministic) ----
  cvt(w_c1, Wc1b, 131072);  cvt(w_c5, Wc5b, 131072);
  cvt(w_re, Wreb, 65536);   cvt(w_im, Wimb, 65536);
  cvt(w_c2, Wc2b, 65536);   cvt(w_c3, Wc3b, 65536);
  cvt(w_c4, Wc4b, 65536);   cvt(w_c6, Wc6b, 65536);
  cvt(w_c7, Wc7b, 65536);   cvt(w_ca, Wcab, 65536);
  cvt(w_fg, Wfgb, 65536);   cvt(w_q,  Wqb,  65536);
  cvt(w_k,  Wkb,  65536);   cvt(w_v,  Wvb,  65536);
  cvt(w_o,  Wob,  65536);

  // ---- pipeline ----
  layernorm(event, S[0]);                                   // x1
  layernorm(image, S[1]);                                   // x2
  conv(S[0], S[1], Wc1b, b_c1, S[2], 512, 0);               // x3
  dw(S[2], w_dw1, b_dw1, S[3], 0);
  zero(S[4]);
  fft_rows(S[3], S[4], -1.f, 1.f); fft_cols(S[3], S[4], -1.f, 1.f);  // x4
  conv(S[3], nullptr, Wreb, b_re, S[5], 256, 1);            // real
  conv(S[4], nullptr, Wimb, b_im, S[6], 256, 1);            // imag
  conv(S[5], S[6], Wc5b, b_c5, S[3], 512, 2);               // x5 (sigmoid)
  conv(S[1], nullptr, Wc2b, b_c2, S[4], 256, 0);
  dw(S[4], w_dw2, b_dw2, S[5], 0);
  zero(S[6]);
  fft_rows(S[5], S[6], -1.f, 1.f); fft_cols(S[5], S[6], -1.f, 1.f);  // x6
  ew_mul_kernel<<<EWG, 256, 0, stream>>>(S[3], S[5], S[4]); // (x5*x6).real
  conv(S[4], nullptr, Wc4b, b_c4, S[5], 256, 1);            // relu(c4)
  conv(S[5], nullptr, Wc3b, b_c3, S[3], 256, 0);            // c3
  zero(S[6]);
  fft_rows(S[3], S[6], 1.f, IS); fft_cols(S[3], S[6], 1.f, IS);      // x7 = S3
  ew_copy_kernel<<<EWG, 256, 0, stream>>>(S[3], S[4]);
  zero(S[5]);
  fft_rows(S[4], S[5], -1.f, 1.f);                          // x8 = (S4,S5)
  ew_add_kernel<<<EWG, 256, 0, stream>>>(S[3], S[2], S[6]); // x7 + x3
  conv(S[6], nullptr, Wcab, b_ca, S[7], 256, 0);
  conv(S[7], nullptr, Wfgb, b_fg, S[6], 256, 0);            // x9 = S6
  ew_mul2_kernel<<<EWG, 256, 0, stream>>>(S[4], S[5], S[6]);// x8 * x9
  fft_rows(S[4], S[5], 1.f, IS);                            // x10 = S4 (real)
  conv(S[0], nullptr, Wc6b, b_c6, S[2], 256, 0);            // kv (x1 dead after)
  conv(S[4], nullptr, Wqb, b_q, S[5], 256, 0);              // Q
  conv(S[2], nullptr, Wkb, b_k, S[6], 256, 0);              // K
  conv(S[2], nullptr, Wvb, b_v, S[7], 256, 0);              // V
  attn4_kernel<<<2048, 256, 0, stream>>>(S[5], S[6], S[7], S[0]);
  conv(S[0], nullptr, Wob, b_o, S[4], 256, 0);              // xc
  dw(S[4], w_dw3, b_dw3, S[5], 1);                          // gelu(dw3)
  conv(S[5], nullptr, Wc7b, b_c7, S[6], 256, 0);            // c7
  ew_add_kernel<<<EWG, 256, 0, stream>>>(S[6], S[1], (float*)d_out); // + x2
}
